// QConv2d_24034636988577
// MI455X (gfx1250) — matrix operand order repaired and checked by reading, not compiled, not run on hardware
//
#include <hip/hip_runtime.h>

// ---------------------------------------------------------------------------
// QConv2d (bf16-quantized 3x3 conv, fp32 accumulate) as implicit GEMM on
// CDNA5 WMMA:  D[n][m] += Wq(16x32 bf16) x Aq(32x16 bf16)  via
// v_wmma_f32_16x16x32_bf16.   M = 32*3136, N = 128, K = 576 = 18 x 32.
// Weight tiles staged with gfx1250 async global->LDS copies when available.
// ---------------------------------------------------------------------------

typedef __bf16 bf16_t;
typedef __attribute__((ext_vector_type(16))) __bf16 v16bf;
typedef __attribute__((ext_vector_type(8)))  __bf16 v8bf;
typedef __attribute__((ext_vector_type(8)))  float   v8f;
typedef __attribute__((ext_vector_type(4)))  int     v4i;

#define IMG_N   32
#define CIN     64
#define HW      56
#define PHW     58          // 56 + 2 (zero padding baked in)
#define NOUT    128
#define KTOT    576         // 64 * 9, reordered as (kh,kw,c)
#define KSTEP   32
#define NSTEPS  18
#define MPI     3136        // 56*56 rows per image
#define MTILE   64
#define MBLKS   49          // 3136 / 64
#define LDA     40          // LDS row stride in halves (32 data + 8 pad = 80 B)

// workspace layout (bytes); total need ~= 13.3 MB
#define WQ_OFF   0                          // bf16[128*576] reordered weights
#define QB_OFF   (NOUT * KTOT * 2)          // float[128] quantized bias
#define PIN_OFF  (QB_OFF + 512)             // bf16[32*64*58*58] padded input

#if __has_builtin(__builtin_amdgcn_global_load_async_to_lds_b128) && \
    __has_builtin(__builtin_amdgcn_s_wait_asynccnt)
#define HAVE_ASYNC_LDS 1
#else
#define HAVE_ASYNC_LDS 0
#endif

#define AS1 __attribute__((address_space(1)))
#define AS3 __attribute__((address_space(3)))

// --------------------------- prepack: weights + bias ------------------------
__global__ __launch_bounds__(256) void prep_weights(const float* __restrict__ w,
                                                    const float* __restrict__ bias,
                                                    bf16_t* __restrict__ wq,
                                                    float* __restrict__ qb) {
    int idx = blockIdx.x * 256 + threadIdx.x;
    if (idx < NOUT * KTOT) {
        int n   = idx / KTOT;
        int r   = idx - n * KTOT;     // src k = c*9 + kh*3 + kw
        int c   = r / 9;
        int khw = r - c * 9;
        // dst k' = (kh*3+kw)*64 + c  -> K-step s covers bytes [s*64, s*64+64)
        wq[n * KTOT + khw * CIN + c] = (bf16_t)w[idx];   // RNE f32->bf16 quant
    }
    if (idx < NOUT) qb[idx] = (float)(bf16_t)bias[idx];  // quantize-dequantize
}

// --------------------- prepack: zero-padded bf16 input ----------------------
__global__ __launch_bounds__(256) void prep_input(const float* __restrict__ x,
                                                  bf16_t* __restrict__ pin) {
    int idx = blockIdx.x * 256 + threadIdx.x;
    if (idx >= IMG_N * CIN * PHW * PHW) return;
    int cg = idx / (PHW * PHW);             // (img*64 + c)
    int r  = idx - cg * (PHW * PHW);
    int h  = r / PHW;
    int ww = r - h * PHW;
    bf16_t v = (bf16_t)0.0f;
    if (h >= 1 && h <= HW && ww >= 1 && ww <= HW)
        v = (bf16_t)x[(cg * HW + (h - 1)) * HW + (ww - 1)];
    pin[idx] = v;
}

// ------------------------------ main GEMM kernel ----------------------------
__global__ __launch_bounds__(256, 2) void qconv_wmma(const bf16_t* __restrict__ pin,
                                                     const bf16_t* __restrict__ wq,
                                                     const float*  __restrict__ qb,
                                                     float*        __restrict__ out) {
    __shared__ __attribute__((aligned(16))) bf16_t As[2][MTILE * LDA];  // acts  [m][k]
    __shared__ __attribute__((aligned(16))) bf16_t Ws[2][NOUT  * LDA];  // wghts [n][k]

    const int tid  = threadIdx.x;
    const int img  = blockIdx.x / MBLKS;
    const int m0   = (blockIdx.x - img * MBLKS) * MTILE;

    const int lane  = tid & 31;          // wave32
    const int wv    = tid >> 5;          // 8 waves
    const int lo    = lane & 15;
    const int hi    = lane >> 4;
    const int mtile = wv & 3;            // which 16-row m strip
    const int n0    = (wv >> 2) * 64;    // which 64-col n half

    // staging coordinates for activations (64 rows x 32 channels per step)
    const int ar  = tid & 63;            // m-local row
    const int acq = tid >> 6;            // channel octet 0..3
    const int am  = m0 + ar;
    const int aoh = am / HW;
    const int aow = am - aoh * HW;

    v8f acc[4] = {};                     // 4 n-tiles of 16x16 f32

    auto stage = [&](int s, int buf) {
        const int khw = s >> 1;
        const int c0  = (s & 1) * 32;
        const int kh  = khw / 3;
        const int kw  = khw - kh * 3;
        // weights: contiguous 64B per row -> two 16B chunks per thread.
        // Pure bf16 copy: use async global->LDS DMA path when available so the
        // transfer overlaps WMMA on the other buffer (ASYNCcnt-tracked).
#pragma unroll
        for (int it = 0; it < 2; ++it) {
            int q = tid + it * 256;
            int n = q >> 2;
            int part = q & 3;
            const bf16_t* g = wq + (size_t)n * KTOT + s * KSTEP + part * 8;
            bf16_t*      l = &Ws[buf][n * LDA + part * 8];
#if HAVE_ASYNC_LDS
            __builtin_amdgcn_global_load_async_to_lds_b128(
                (AS1 v4i*)g, (AS3 v4i*)l, 0, 0);
#else
            *(v8bf*)l = *(const v8bf*)g;
#endif
        }
        // activations: As[r][c] = pin[img, c0+c, aoh+kh, aow+kw]  (never OOB).
        // 8 consecutive channels per thread -> one 16B ds_store_b128.
        const bf16_t* src = pin + (((size_t)(img * CIN + c0 + acq * 8) * PHW)
                                   + (aoh + kh)) * PHW + (aow + kw);
        union { v8bf v; bf16_t e[8]; } arow;
#pragma unroll
        for (int j = 0; j < 8; ++j)
            arow.e[j] = src[(size_t)j * (PHW * PHW)];
        *(v8bf*)&As[buf][ar * LDA + acq * 8] = arow.v;
    };

    auto compute = [&](int buf) {
        const bf16_t* Ab = &As[buf][0];
        const bf16_t* Wb = &Ws[buf][0];
        // B operand (acts, 32x16): lane lo = column m, K = {hi*8.., 16+hi*8..}
        union { v16bf v; v8bf h[2]; } bfrag;
        const int abase = (mtile * 16 + lo) * LDA + hi * 8;
        bfrag.h[0] = *(const v8bf*)(Ab + abase);
        bfrag.h[1] = *(const v8bf*)(Ab + abase + 16);
#pragma unroll
        for (int j = 0; j < 4; ++j) {
            // A operand (weights, 16x32): lane lo = row n, K = {hi*8.., 16+hi*8..}
            union { v16bf v; v8bf h[2]; } afrag;
            const int wbase = (n0 + j * 16 + lo) * LDA + hi * 8;
            afrag.h[0] = *(const v8bf*)(Wb + wbase);
            afrag.h[1] = *(const v8bf*)(Wb + wbase + 16);
            acc[j] = __builtin_amdgcn_wmma_f32_16x16x32_bf16(
                false, afrag.v, false, bfrag.v, (short)0, acc[j], false, false);
        }
    };

    stage(0, 0);
#if HAVE_ASYNC_LDS
    __builtin_amdgcn_s_wait_asynccnt(0);
#endif
    __syncthreads();
#pragma unroll 1
    for (int s = 0; s < NSTEPS; ++s) {
        const int cur = s & 1;
        if (s + 1 < NSTEPS) stage(s + 1, cur ^ 1);   // fill other buffer (async)
        compute(cur);                                // WMMA on current buffer
#if HAVE_ASYNC_LDS
        __builtin_amdgcn_s_wait_asynccnt(0);
#endif
        __syncthreads();
    }

    // epilogue: D lane dim = m (contiguous ow) -> coalesced 64B store runs.
    // Bias values for one tile are 8 contiguous floats -> one 32B load + packed add.
    const int m = m0 + mtile * 16 + lo;
#pragma unroll
    for (int j = 0; j < 4; ++j) {
        const int nb = n0 + j * 16 + 8 * hi;
        const v8f qbv = *(const v8f*)(qb + nb);
        const v8f r = acc[j] + qbv;
#pragma unroll
        for (int v = 0; v < 8; ++v)
            out[((size_t)(img * NOUT + nb + v)) * MPI + m] = r[v];
    }
}

// ------------------------------------ launch --------------------------------
extern "C" void kernel_launch(void* const* d_in, const int* in_sizes, int n_in,
                              void* d_out, int out_size, void* d_ws, size_t ws_size,
                              hipStream_t stream) {
    const float* x  = (const float*)d_in[0];   // [32,64,56,56]
    const float* w  = (const float*)d_in[1];   // [128,64,3,3]
    const float* bs = (const float*)d_in[2];   // [128]
    float* out = (float*)d_out;                // [32,128,56,56]

    char* ws = (char*)d_ws;                    // needs ~13.3 MB scratch
    bf16_t* wq  = (bf16_t*)(ws + WQ_OFF);
    float*  qb  = (float*)(ws + QB_OFF);
    bf16_t* pin = (bf16_t*)(ws + PIN_OFF);

    prep_weights<<<(NOUT * KTOT + 255) / 256, 256, 0, stream>>>(w, bs, wq, qb);
    const int pelems = IMG_N * CIN * PHW * PHW;
    prep_input<<<(pelems + 255) / 256, 256, 0, stream>>>(x, pin);
    qconv_wmma<<<IMG_N * MBLKS, 256, 0, stream>>>(pin, wq, qb, out);
}
